// InterpNet_59365037965877
// MI455X (gfx1250) — matrix operand order repaired and checked by reading, not compiled, not run on hardware
//
#include <hip/hip_runtime.h>

// ---------------------------------------------------------------------------
// InterpNet edge MLP for MI455X (gfx1250, wave32, WMMA).
//
//   x = concat(latents[col], pos_target[row]-pos_source[col])   [E, 259]
//   x = x @ W_in + b_in ; x = relu(x) @ W1 + b1 ; x = relu(x) @ W2 + b2
//   out = (x @ W_out + b_out)[:, 0]
//
// bf16 WMMA (v_wmma_f32_16x16x32_bf16), f32 accumulate. Weights packed once
// to bf16 in d_ws (W_in K padded 259->288). One wave32 per 48-edge tile
// (3 M-tiles); N split into 4 groups of 64 cols so each 1KB B-fragment is
// reused by 3 WMMAs. B-fragments are software-pipelined one K-step ahead to
// hide L2 latency at low occupancy. Activations ping-pong between two
// per-wave LDS buffers (1 wave/block, barrier-free).
// ---------------------------------------------------------------------------

typedef __attribute__((ext_vector_type(16))) __bf16 bf16x16;
typedef __attribute__((ext_vector_type(8)))  __bf16 bf16x8;
typedef __attribute__((ext_vector_type(8)))  float  v8f;

#define LFEAT   256
#define KPAD    288              // 259 padded to 9 K-tiles of 32
#define STRIDE0 296              // act0 row stride (bf16), 16B-aligned rows
#define STRIDE1 264              // act1 row stride (bf16)
#define MTILES  3
#define MEDGE   (MTILES * 16)    // 48 edges per wave

// bf16 workspace layout (element offsets)
#define WS_WIN   0                        // [288][256]
#define WS_W1    (KPAD * LFEAT)           // [256][256]
#define WS_W2    (WS_W1 + LFEAT * LFEAT)  // [256][256]
#define WS_TOTAL (WS_W2 + LFEAT * LFEAT)  // 204800 bf16 = 400 KB

// ---------------------------------------------------------------------------
__global__ void pack_weights_kernel(const float* __restrict__ Win,
                                    const float* __restrict__ W1,
                                    const float* __restrict__ W2,
                                    __bf16* __restrict__ wbuf) {
    int i = blockIdx.x * blockDim.x + threadIdx.x;
    int base = (i < WS_W1) ? WS_WIN : (i < WS_W2) ? WS_W1 : WS_W2;
    int r = (i - base) >> 8, c = (i - base) & 255;
    int L = r & 31, h = L >> 4, s = c & 15;
    int k = (r & ~31) + ((s < 8) ? 8 * h + s : 16 + 8 * h + (s - 8));
    int n = (c & ~15) | (L & 15);
    if (i < KPAD * LFEAT) {
        wbuf[i] = (k < LFEAT + 3) ? (__bf16)Win[k * LFEAT + n] : (__bf16)0.0f;
    } else if (i < WS_W2) {
        wbuf[i] = (__bf16)W1[k * LFEAT + n];
    } else if (i < WS_TOTAL) {
        wbuf[i] = (__bf16)W2[k * LFEAT + n];
    }
}

// ---------------------------------------------------------------------------
// One layer pass: for each of 4 N-groups (64 cols), sweep K; B frags are
// loaded once per (kt, ngroup), reused across 3 M-tiles, and pipelined one
// kt ahead so the 12 WMMAs of step kt hide the L2 latency of step kt+1.
// A-frag layout (ISA 7.12.2, 16-bit A 16x32):
//   lanes 0-15 : M=lane,    K = k0+0..7  / k0+16..23
//   lanes 16-31: M=lane-16, K = k0+8..15 / k0+24..31
// B-frag: lane L = N-col (L&15), 16 contiguous bf16 along K in A's K order (packed).
// C layout: VGPR r, lane l -> M = r + 8*(l>=16), N = l&15.
// ---------------------------------------------------------------------------
template<int KTILES, bool RELU_STORE>
__device__ __forceinline__ void mlp_layer(const __bf16* __restrict__ Wb,
                                          const float*  __restrict__ bias,
                                          const __bf16* in_act, int in_stride,
                                          __bf16* out_act, int out_stride,
                                          const float* __restrict__ W_out,
                                          float (*pacc)[8],
                                          int lane) {
    const int half = lane >> 4;
    const int nl   = lane & 15;
    const int mrow = lane & 15;

    #pragma unroll 1
    for (int ng = 0; ng < 4; ++ng) {
        v8f c[MTILES][4];
        #pragma unroll
        for (int j = 0; j < 4; ++j) {
            float bv = bias[ng * 64 + j * 16 + nl];
            #pragma unroll
            for (int mt = 0; mt < MTILES; ++mt) {
                #pragma unroll
                for (int i = 0; i < 8; ++i) c[mt][j][i] = bv;
            }
        }

        // --- B pipeline prologue: fragments for kt = 0 ---
        const __bf16* wrow0 = Wb + (size_t)lane * LFEAT + ng * 64;
        bf16x16 bcur[4];
        #pragma unroll
        for (int j = 0; j < 4; ++j)
            bcur[j] = *(const bf16x16*)(wrow0 + j * 16);

        #pragma unroll 2
        for (int kt = 0; kt < KTILES; ++kt) {
            // A-fragments for this kt (separate regs -> all ds loads issue
            // before the first WMMA, no serialization on dscnt)
            const int ka = kt * 32 + half * 8;
            bf16x16 a[MTILES];
            #pragma unroll
            for (int mt = 0; mt < MTILES; ++mt) {
                const __bf16* arow = in_act + (size_t)(mt * 16 + mrow) * in_stride;
                bf16x8 alo = *(const bf16x8*)(arow + ka);
                bf16x8 ahi = *(const bf16x8*)(arow + ka + 16);
                a[mt] = __builtin_shufflevector(alo, ahi,
                            0,1,2,3,4,5,6,7,8,9,10,11,12,13,14,15);
            }

            // B-fragments for kt+1 (clamped; redundant reload on last kt)
            const int ktn = (kt + 1 < KTILES) ? kt + 1 : kt;
            const int ktp = (kt + 2 < KTILES) ? kt + 2 : KTILES - 1;
            const __bf16* wnext = Wb + (size_t)(ktn * 32 + lane) * LFEAT + ng * 64;
            __builtin_prefetch(Wb + (size_t)(ktp * 32 + lane) * LFEAT + ng * 64, 0, 1);
            bf16x16 bnxt[4];
            #pragma unroll
            for (int j = 0; j < 4; ++j)
                bnxt[j] = *(const bf16x16*)(wnext + j * 16);

            // 12 WMMAs on the already-resident bcur
            #pragma unroll
            for (int mt = 0; mt < MTILES; ++mt)
                #pragma unroll
                for (int j = 0; j < 4; ++j)
                    c[mt][j] = __builtin_amdgcn_wmma_f32_16x16x32_bf16(
                                   false, a[mt], false, bcur[j],
                                   (short)0, c[mt][j], false, false);

            #pragma unroll
            for (int j = 0; j < 4; ++j) bcur[j] = bnxt[j];  // rotated by unroll
        }

        if constexpr (RELU_STORE) {
            #pragma unroll
            for (int mt = 0; mt < MTILES; ++mt)
                #pragma unroll
                for (int j = 0; j < 4; ++j)
                    #pragma unroll
                    for (int r = 0; r < 8; ++r) {
                        float v = fmaxf(c[mt][j][r], 0.0f);
                        out_act[(size_t)(mt * 16 + half * 8 + r) * out_stride
                                + ng * 64 + j * 16 + nl] = (__bf16)v;
                    }
        } else {
            #pragma unroll
            for (int j = 0; j < 4; ++j) {
                float wv = W_out[(size_t)(ng * 64 + j * 16 + nl) * 2]; // col 0
                #pragma unroll
                for (int mt = 0; mt < MTILES; ++mt)
                    #pragma unroll
                    for (int r = 0; r < 8; ++r)
                        pacc[mt][r] += c[mt][j][r] * wv;
            }
        }
    }
}

// ---------------------------------------------------------------------------
// Main fused kernel: one wave32 per block, 48 edges per wave.
// ---------------------------------------------------------------------------
__global__ __launch_bounds__(32)
void interp_edge_mlp(const float* __restrict__ pos_source,
                     const float* __restrict__ pos_target,
                     const float* __restrict__ latents,
                     const int*   __restrict__ row,
                     const int*   __restrict__ col,
                     const __bf16* __restrict__ wbuf,
                     const float* __restrict__ b_in,
                     const float* __restrict__ b1,
                     const float* __restrict__ b2,
                     const float* __restrict__ W_out,
                     const float* __restrict__ b_out,
                     float* __restrict__ out,
                     int E) {
    __shared__ __align__(32) __bf16 act0[MEDGE][STRIDE0];  // K=288 / hidden
    __shared__ __align__(32) __bf16 act1[MEDGE][STRIDE1];  // hidden (K=256)

    const int lane = threadIdx.x & 31;
    const long e0  = (long)blockIdx.x * MEDGE;
    if (e0 >= E) return;

    const int half = lane >> 4;
    const int nl   = lane & 15;

    // ---- Gather A0 = [latents[col] | pos_rel | 0-pad] as bf16 in LDS ----
    // col[e] is wave-uniform per m -> scalar load; lanes fetch 32B each.
    for (int m = 0; m < MEDGE; ++m) {
        long e = e0 + m; if (e >= E) e = E - 1;
        int cs = col[e];
        const float4* src = (const float4*)(latents + (size_t)cs * LFEAT);
        float4 f0 = src[lane * 2 + 0];
        float4 f1 = src[lane * 2 + 1];
        __bf16* dst = &act0[m][0] + lane * 8;
        dst[0] = (__bf16)f0.x; dst[1] = (__bf16)f0.y;
        dst[2] = (__bf16)f0.z; dst[3] = (__bf16)f0.w;
        dst[4] = (__bf16)f1.x; dst[5] = (__bf16)f1.y;
        dst[6] = (__bf16)f1.z; dst[7] = (__bf16)f1.w;
    }
    for (int m = lane; m < MEDGE; m += 32) {
        long e = e0 + m; if (e >= E) e = E - 1;
        int rs = row[e];
        int cs = col[e];
        float px = pos_target[(size_t)rs * 3 + 0] - pos_source[(size_t)cs * 3 + 0];
        float py = pos_target[(size_t)rs * 3 + 1] - pos_source[(size_t)cs * 3 + 1];
        float pz = pos_target[(size_t)rs * 3 + 2] - pos_source[(size_t)cs * 3 + 2];
        __bf16* dst = &act0[m][0];
        dst[LFEAT + 0] = (__bf16)px;
        dst[LFEAT + 1] = (__bf16)py;
        dst[LFEAT + 2] = (__bf16)pz;
        #pragma unroll
        for (int k = LFEAT + 3; k < KPAD; ++k) dst[k] = (__bf16)0.0f;
    }
    // Single wave per block: in-wave DS ordering suffices, no barrier.

    // ---- Fused MLP (ping-pong activations between act0 and act1) ----
    float pacc[MTILES][8];
    #pragma unroll
    for (int mt = 0; mt < MTILES; ++mt)
        #pragma unroll
        for (int r = 0; r < 8; ++r) pacc[mt][r] = 0.0f;

    mlp_layer<KPAD / 32, true>(wbuf + WS_WIN, b_in,
                               &act0[0][0], STRIDE0, &act1[0][0], STRIDE1,
                               nullptr, nullptr, lane);          // relu(x1)
    mlp_layer<LFEAT / 32, true>(wbuf + WS_W1, b1,
                                &act1[0][0], STRIDE1, &act0[0][0], STRIDE0,
                                nullptr, nullptr, lane);         // relu(x2)
    mlp_layer<LFEAT / 32, false>(wbuf + WS_W2, b2,
                                 &act0[0][0], STRIDE0, nullptr, 0,
                                 W_out, pacc, lane);  // x3 . W_out[:,0] partial

    // ---- Reduce the N dimension across each 16-lane half, then store ----
    #pragma unroll
    for (int mt = 0; mt < MTILES; ++mt)
        #pragma unroll
        for (int r = 0; r < 8; ++r) {
            float v = pacc[mt][r];
            v += __shfl_xor(v, 1, 16);
            v += __shfl_xor(v, 2, 16);
            v += __shfl_xor(v, 4, 16);
            v += __shfl_xor(v, 8, 16);
            pacc[mt][r] = v;
        }
    if (nl == 0) {
        float bo = b_out[0];
        #pragma unroll
        for (int mt = 0; mt < MTILES; ++mt)
            #pragma unroll
            for (int r = 0; r < 8; ++r) {
                long e = e0 + mt * 16 + half * 8 + r;
                if (e < E) out[e] = pacc[mt][r] + bo;
            }
    }
}

// ---------------------------------------------------------------------------
extern "C" void kernel_launch(void* const* d_in, const int* in_sizes, int n_in,
                              void* d_out, int out_size, void* d_ws, size_t ws_size,
                              hipStream_t stream) {
    const float* pos_source = (const float*)d_in[0];
    const float* pos_target = (const float*)d_in[1];
    const float* latents    = (const float*)d_in[2];
    const int*   row        = (const int*)  d_in[3];
    const int*   col        = (const int*)  d_in[4];
    const float* W_in       = (const float*)d_in[5];
    const float* b_in       = (const float*)d_in[6];
    const float* W1         = (const float*)d_in[7];
    const float* b1         = (const float*)d_in[8];
    const float* W2         = (const float*)d_in[9];
    const float* b2         = (const float*)d_in[10];
    const float* W_out      = (const float*)d_in[11];
    const float* b_out      = (const float*)d_in[12];

    const int E = in_sizes[3];
    __bf16* wbuf = (__bf16*)d_ws;   // needs WS_TOTAL*2 = 400 KB of d_ws

    // 1) pack weights to bf16 (stream-ordered; main kernel waits in-stream)
    {
        int blocks = (WS_TOTAL + 255) / 256;
        pack_weights_kernel<<<blocks, 256, 0, stream>>>(W_in, W1, W2, wbuf);
    }

    // 2) fused edge MLP: one wave32 per 48-edge tile
    {
        int nblocks = (E + MEDGE - 1) / MEDGE;
        interp_edge_mlp<<<nblocks, 32, 0, stream>>>(
            pos_source, pos_target, latents, row, col,
            wbuf, b_in, b1, b2, W_out, b_out,
            (float*)d_out, E);
    }
}
